// QuestionDecoder_35390530519496
// MI455X (gfx1250) — matrix operand order repaired and checked by reading, not compiled, not run on hardware
//
#include <hip/hip_runtime.h>
#include <hip/hip_bf16.h>
#include <math.h>
#include <stdint.h>

// ---------------- model constants ----------------
#define VV   30522
#define HH   768
#define NZp  2
#define LL   2
#define NHp  8
#define BB   8
#define TCp  384
#define TQp  64

typedef __attribute__((ext_vector_type(16))) __bf16          v16bf;
typedef __attribute__((ext_vector_type(16))) unsigned short  v16u;
typedef __attribute__((ext_vector_type(8)))  float           v8f;
typedef __attribute__((ext_vector_type(4)))  unsigned int    v4u;
typedef __attribute__((ext_vector_type(8)))  int             v8i;
typedef __attribute__((ext_vector_type(4)))  int             v4i;

#if defined(__HIP_DEVICE_COMPILE__) && __has_builtin(__builtin_amdgcn_tensor_load_to_lds)
#define USE_TDM 1
#else
#define USE_TDM 0
#endif

// ---------------- helpers ----------------
__device__ __forceinline__ unsigned short f2bf(float f) {
    unsigned int u = __builtin_bit_cast(unsigned int, f);
    u += 0x7FFFu + ((u >> 16) & 1u);      // round-to-nearest-even
    return (unsigned short)(u >> 16);
}

__device__ __forceinline__ float act_apply(int act, float x) {
    switch (act) {
        case 1: return 0.5f * x * (1.0f + erff(x * 0.70710678118654752f));    // exact gelu
        case 2: return tanhf(x);
        case 3: return 1.0f / (1.0f + __expf(-x));
        case 4: { float sp = (x > 20.0f) ? x : log1pf(__expf(x));             // mish
                  return x * tanhf(sp); }
        default: return x;
    }
}

// block reductions (blockDim.x == 128)
__device__ __forceinline__ float bred_sum(float v, float* sh) {
    int t = threadIdx.x;
    __syncthreads(); sh[t] = v; __syncthreads();
    for (int s = 64; s > 0; s >>= 1) { if (t < s) sh[t] += sh[t + s]; __syncthreads(); }
    return sh[0];
}
__device__ __forceinline__ float bred_max(float v, float* sh) {
    int t = threadIdx.x;
    __syncthreads(); sh[t] = v; __syncthreads();
    for (int s = 64; s > 0; s >>= 1) { if (t < s) sh[t] = fmaxf(sh[t], sh[t + s]); __syncthreads(); }
    return sh[0];
}

#if USE_TDM
// Issue one 2D TDM tile load: global (row-major, stride `stride_elems` f32 per row)
// -> LDS at byte offset lds_off. tile_w f32 contiguous per row, tile_h rows.
// rem_w / rem_h are the remaining tensor extents from the tile origin; TDM
// zero-fills reads beyond them (hardware OOB handling for edge tiles).
__device__ __forceinline__ void tdm_load_2d(unsigned lds_off, const float* gaddr,
                                            int tile_w, int tile_h,
                                            long long rem_w, long long rem_h,
                                            long long stride_elems)
{
    unsigned long long ga = (unsigned long long)(uintptr_t)gaddr;
    unsigned td0 = (unsigned)(rem_w < 0 ? 0 : rem_w);
    unsigned td1 = (unsigned)(rem_h < 0 ? 0 : rem_h);
    unsigned long long st0 = (unsigned long long)stride_elems;

    v4u g0;
    g0[0] = 1u;                                              // count=1 (valid user D#)
    g0[1] = lds_off;                                         // LDS byte address
    g0[2] = (unsigned)(ga & 0xffffffffull);                  // global_addr[31:0]
    g0[3] = (unsigned)((ga >> 32) & 0x01ffffffull)           // global_addr[56:32]
          | (2u << 30);                                      // type = 2 ("image")

    v8i g1;
    g1[0] = (int)(2u << 16);                                 // data_size = 4B; mask=0
    g1[1] = (int)((td0 & 0xffffu) << 16);                    // tensor_dim0[15:0]
    g1[2] = (int)((td0 >> 16) | ((td1 & 0xffffu) << 16));    // td0 hi | td1 lo
    g1[3] = (int)((td1 >> 16) | ((unsigned)tile_w << 16));   // td1 hi | tile_dim0
    g1[4] = (int)((unsigned)tile_h & 0xffffu);               // tile_dim1; tile_dim2=0
    g1[5] = (int)(st0 & 0xffffffffull);                      // tensor_dim0_stride lo
    g1[6] = (int)((st0 >> 32) & 0xffffull);                  // stride hi | dim1_stride lo=0
    g1[7] = 0;

    v4i z4 = {0, 0, 0, 0};
    v8i z8 = {0, 0, 0, 0, 0, 0, 0, 0};
    __builtin_amdgcn_tensor_load_to_lds(g0, g1, z4, z4, z8, 0);
}

__device__ __forceinline__ unsigned lds_byte_off(const void* p) {
    // generic pointer to LDS: low 32 bits carry the LDS byte offset
    return (unsigned)(unsigned long long)(uintptr_t)p;
}
#endif

// ---------------- WMMA GEMM ----------------
// C[m,n] = act(sum_k A[m,k]*Bop[k,n] + bias[n]) + res[m,n]
// bKN==0 : B stored [N,K] row-major (out = A @ B^T)  -- weight layout
// bKN==1 : B stored [K,N] row-major (out = A @ B)
// batched via blockIdx.z = b*nh + h with per-b / per-h element strides.
// Staging: Tensor Data Mover DMAs raw f32 tiles into LDS (wave0: A, wave1: B),
// fragments are built with in-register f32->bf16 conversion.
__global__ __launch_bounds__(128)
void k_gemm(const float* __restrict__ A, const float* __restrict__ B,
            const float* __restrict__ bias, const float* __restrict__ res,
            float* __restrict__ C,
            int M, int N, int K, int lda, int ldb, int ldc,
            int bKN, int act, int nh,
            long long sAb, long long sAh, long long sBb, long long sBh,
            long long sCb, long long sCh)
{
    __shared__ float Asf[64 * 32];     // A tile, [m][k] 64x32
    __shared__ float Bsf[64 * 32];     // B tile: NK -> [n][k] 64x32 ; KN -> [k][n] 32x64

    int bz = blockIdx.z;
    int bb = bz / nh, hh = bz % nh;
    const float* Ap = A + bb * sAb + hh * sAh;
    const float* Bp = B + bb * sBb + hh * sBh;
    float*       Cp = C + bb * sCb + hh * sCh;
    const float* Rp = res ? (res + bb * sCb + hh * sCh) : nullptr;

    int m0 = blockIdx.y * 64;
    int n0 = blockIdx.x * 64;

    int lane = threadIdx.x & 31;
    int wave = threadIdx.x >> 5;   // 4 waves, wave handles rows [wave*16, wave*16+16)
    int m16  = lane & 15;
    int hi   = lane >> 4;

    v8f zero = {0.f, 0.f, 0.f, 0.f, 0.f, 0.f, 0.f, 0.f};
    v8f acc[4];
#pragma unroll
    for (int i = 0; i < 4; ++i) acc[i] = zero;

    for (int k0 = 0; k0 < K; k0 += 32) {
#if USE_TDM
        // ---- async tile staging via Tensor Data Mover ----
        if (wave == 0) {
            tdm_load_2d(lds_byte_off(Asf), Ap + (long long)m0 * lda + k0,
                        32, 64, (long long)K - k0, (long long)M - m0, lda);
        } else if (wave == 1) {
            if (bKN) {
                // natural [K][N] tile: 32 rows of 64 contiguous n
                tdm_load_2d(lds_byte_off(Bsf), Bp + (long long)k0 * ldb + n0,
                            64, 32, (long long)N - n0, (long long)K - k0, ldb);
            } else {
                // weight [N,K] tile: 64 rows of 32 contiguous k
                tdm_load_2d(lds_byte_off(Bsf), Bp + (long long)n0 * ldb + k0,
                            32, 64, (long long)K - k0, (long long)N - n0, ldb);
            }
        }
        __builtin_amdgcn_s_wait_tensorcnt(0);   // issuing waves drain their TDM
        __syncthreads();                        // publish LDS to all waves
#else
        // ---- fallback: manual staged copy ----
        for (int i = threadIdx.x; i < 64 * 32; i += 128) {
            int r = i >> 5, c = i & 31;
            int gm = m0 + r, gk = k0 + c;
            Asf[i] = (gm < M && gk < K) ? Ap[(long long)gm * lda + gk] : 0.0f;
        }
        for (int i = threadIdx.x; i < 64 * 32; i += 128) {
            float v;
            if (bKN) {
                int r = i >> 6, c = i & 63;          // [k][n]
                int gk = k0 + r, gn = n0 + c;
                v = (gn < N && gk < K) ? Bp[(long long)gk * ldb + gn] : 0.0f;
            } else {
                int r = i >> 5, c = i & 31;          // [n][k]
                int gn = n0 + r, gk = k0 + c;
                v = (gn < N && gk < K) ? Bp[(long long)gn * ldb + gk] : 0.0f;
            }
            Bsf[i] = v;
        }
        __syncthreads();
#endif

        // A fragment: row = wave*16+m16; lane<16 => K 0..15, lane>=16 => K 16..31
        v16u au;
        int ka = hi * 16;
        int ar = wave * 16 + m16;
#pragma unroll
        for (int e = 0; e < 16; ++e) {
            au[e]     = f2bf(Asf[ar * 32 + ka + e]);
        }
        v16bf af = __builtin_bit_cast(v16bf, au);

#pragma unroll
        for (int nt = 0; nt < 4; ++nt) {
            // B fragment: col n = nt*16+m16; lane<16 => K 0..15, lane>=16 => K 16..31
            v16u bu;
            int kb = hi * 16;
            int br = nt * 16 + m16;
            if (bKN) {
#pragma unroll
                for (int e = 0; e < 16; ++e) bu[e] = f2bf(Bsf[(kb + e) * 64 + br]);
            } else {
#pragma unroll
                for (int e = 0; e < 16; ++e) bu[e] = f2bf(Bsf[br * 32 + kb + e]);
            }
            v16bf bfv = __builtin_bit_cast(v16bf, bu);
            acc[nt] = __builtin_amdgcn_wmma_f32_16x16x32_bf16(
                false, af, false, bfv, (short)0, acc[nt], false, false);
        }
        __syncthreads();
    }

    // writeback: elem r -> row = wave*16 + hi*8 + r, col = nt*16 + m16
#pragma unroll
    for (int nt = 0; nt < 4; ++nt) {
#pragma unroll
        for (int r = 0; r < 8; ++r) {
            int gm = m0 + wave * 16 + hi * 8 + r;
            int gn = n0 + nt * 16 + m16;
            if (gm < M && gn < N) {
                float v = acc[nt][r];
                if (bias) v += bias[gn];
                v = act_apply(act, v);
                if (Rp) v += Rp[(long long)gm * ldc + gn];
                Cp[(long long)gm * ldc + gn] = v;
            }
        }
    }
}

// ---------------- attention softmax (encoder) ----------------
// rows = B*NH*S, layout [B,NH,S,S]; pad keys where ids[b,k]==0 -> -1e9; in-place
__global__ __launch_bounds__(128)
void k_softmax_enc(float* sc, const int* __restrict__ ids, int NHn, int S, float scale)
{
    __shared__ float sh[128];
    int row = blockIdx.x;
    int bh  = row / S;
    int b   = bh / NHn;
    float* p = sc + (long long)row * S;
    const int* idb = ids + (long long)b * S;

    float mx = -3.4e38f;
    for (int k = threadIdx.x; k < S; k += 128) {
        float v = p[k] * scale;
        if (idb[k] == 0) v = -1e9f;
        p[k] = v;
        mx = fmaxf(mx, v);
    }
    mx = bred_max(mx, sh);
    float sum = 0.f;
    for (int k = threadIdx.x; k < S; k += 128) { float e = __expf(p[k] - mx); p[k] = e; sum += e; }
    sum = bred_sum(sum, sh);
    float inv = 1.0f / sum;
    for (int k = threadIdx.x; k < S; k += 128) p[k] *= inv;
}

// ---------------- cal_attn mask + softmax ----------------
__global__ __launch_bounds__(128)
void k_cal_softmax(float* logits, float* w, const int* __restrict__ cids,
                   const int* __restrict__ qids, int TQn, int TCn)
{
    __shared__ float sh[128];
    int row = blockIdx.x;
    int b = row / TQn, t = row - b * TQn;
    float* lp = logits + (long long)row * TCn;
    float* wp = w      + (long long)row * TCn;
    int qpad = (qids[(long long)b * TQn + t] == 0);
    const int* cb = cids + (long long)b * TCn;

    float mx = -3.4e38f;
    for (int c = threadIdx.x; c < TCn; c += 128) {
        float v = lp[c];
        if (qpad || cb[c] == 0) v = -10000.0f;
        lp[c] = v;
        mx = fmaxf(mx, v);
    }
    mx = bred_max(mx, sh);
    float sum = 0.f;
    for (int c = threadIdx.x; c < TCn; c += 128) { float e = __expf(lp[c] - mx); wp[c] = e; sum += e; }
    sum = bred_sum(sum, sh);
    float inv = 1.0f / sum;
    for (int c = threadIdx.x; c < TCn; c += 128) wp[c] *= inv;
}

// ---------------- layernorm (in place), one row per block ----------------
__global__ __launch_bounds__(128)
void k_layernorm(float* x, const float* __restrict__ g, const float* __restrict__ b, int D)
{
    __shared__ float sh[128];
    float* p = x + (long long)blockIdx.x * D;
    float s = 0.f, s2 = 0.f;
    for (int i = threadIdx.x; i < D; i += 128) { float v = p[i]; s += v; s2 += v * v; }
    float ts  = bred_sum(s,  sh);
    float ts2 = bred_sum(s2, sh);
    float mean = ts / (float)D;
    float var  = ts2 / (float)D - mean * mean;
    float inv  = rsqrtf(var + 1e-5f);
    for (int i = threadIdx.x; i < D; i += 128)
        p[i] = (p[i] - mean) * inv * g[i] + b[i];
}

// ---------------- gathers / concats / elementwise ----------------
__global__ void k_embed_ctx(const int* __restrict__ c_ids, const int* __restrict__ a_ids,
                            const float* __restrict__ emb, float* ce, float* cae,
                            int* ca_ids, int TCn, int Hn)
{
    int row = blockIdx.x;
    int t = row % TCn;
    int cid  = c_ids[row];
    int caid = (t == 0) ? cid : cid * a_ids[row];
    if (threadIdx.x == 0) ca_ids[row] = caid;
    for (int h = threadIdx.x; h < Hn; h += 256) {
        ce [(long long)row * Hn + h] = emb[(long long)cid  * Hn + h];
        cae[(long long)row * Hn + h] = emb[(long long)caid * Hn + h];
    }
}

__global__ void k_decoded_q(const float* __restrict__ zq, const float* __restrict__ zw,
                            const float* __restrict__ zb, float* dq, int Hn)
{
    int i = blockIdx.x * blockDim.x + threadIdx.x;
    if (i >= BB * Hn) return;
    int b = i / Hn, h = i - b * Hn;
    dq[i] = zb[h] + zq[b * NZp + 0] * zw[h * NZp + 0] + zq[b * NZp + 1] * zw[h * NZp + 1];
}

__global__ void k_build_qx(const int* __restrict__ q_ids, const float* __restrict__ emb,
                           const float* __restrict__ dq, float* qx, int TQn, int Hn)
{
    int row = blockIdx.x;
    int b = row / TQn;
    int qid = q_ids[row];
    for (int h = threadIdx.x; h < Hn; h += 256) {
        qx[(long long)row * 2 * Hn + h]      = emb[(long long)qid * Hn + h];
        qx[(long long)row * 2 * Hn + Hn + h] = dq[(long long)b * Hn + h];
    }
}

__global__ void k_concat2(const float* __restrict__ a, const float* __restrict__ b,
                          float* o, int d1, int d2)
{
    long long row = blockIdx.x;
    for (int i = threadIdx.x; i < d1; i += 256) o[row * (d1 + d2) + i]      = a[row * d1 + i];
    for (int i = threadIdx.x; i < d2; i += 256) o[row * (d1 + d2) + d1 + i] = b[row * d2 + i];
}

__global__ void k_gate_combine(const float* __restrict__ g, const float* __restrict__ f,
                               const float* __restrict__ c, float* o, long long n)
{
    long long i = (long long)blockIdx.x * blockDim.x + threadIdx.x;
    if (i < n) o[i] = g[i] * f[i] + (1.0f - g[i]) * c[i];
}

__global__ void k_maxout(const float* __restrict__ qc, float* o, long long n, int Hn)
{
    long long i = (long long)blockIdx.x * blockDim.x + threadIdx.x;
    if (i >= n) return;
    long long m = i / Hn;
    int h = (int)(i - m * Hn);
    float a = qc[m * 2 * Hn + 2 * h];
    float b = qc[m * 2 * Hn + 2 * h + 1];
    o[i] = fmaxf(a, b);
}

// ---------------- pointer-gen copy: scatter-max into vocab, add to out ----------
__global__ __launch_bounds__(TCp)
void k_copy_scatter(float* out, const float* __restrict__ logits,
                    const int* __restrict__ cids, int TQn, int TCn, int Vn)
{
    __shared__ int   sid[TCp];
    __shared__ float sval[TCp];
    int row = blockIdx.x;
    int b = row / TQn;
    int c = threadIdx.x;
    sid[c]  = cids[(long long)b * TCn + c];
    sval[c] = logits[(long long)row * TCn + c];
    __syncthreads();
    int id = sid[c];
    float v = sval[c];
    bool win = true;
    for (int j = 0; j < TCn; ++j) {
        if (sid[j] == id && (sval[j] > v || (sval[j] == v && j < c))) { win = false; break; }
    }
    if (win && v != -10000.0f)
        out[(long long)row * Vn + id] += v;
}

// ---------------- host-side helpers ----------------
static inline void gemm(hipStream_t st, const float* A, const float* B, const float* bias,
                        const float* res, float* C, int M, int N, int K,
                        int lda, int ldb, int ldc, int bKN, int act,
                        int batches = 1, int nh = 1,
                        long long sAb = 0, long long sAh = 0,
                        long long sBb = 0, long long sBh = 0,
                        long long sCb = 0, long long sCh = 0)
{
    dim3 g((N + 63) / 64, (M + 63) / 64, batches);
    k_gemm<<<g, 128, 0, st>>>(A, B, bias, res, C, M, N, K, lda, ldb, ldc,
                              bKN, act, nh, sAb, sAh, sBb, sBh, sCb, sCh);
}

struct EncP {
    const float *qkv_w, *qkv_b, *out_w, *out_b, *ff1_w, *ff1_b, *ff2_w, *ff2_b;
    const float *ln1_g, *ln1_b, *ln2_g, *ln2_b;
};

static void run_encoder(hipStream_t st, float* x, const int* ids,
                        int Bn, int S, int D, int ff, int hd, const EncP& P,
                        float* qkvb, float* scores, float* mha, float* ffb)
{
    int NHn = D / hd;               // == 8
    int M = Bn * S;
    float scale = 1.0f / sqrtf((float)hd);
    for (int l = 0; l < LL; ++l) {
        const float* qw  = P.qkv_w + (long long)l * 3 * D * D;
        const float* qb  = P.qkv_b + (long long)l * 3 * D;
        const float* ow  = P.out_w + (long long)l * D * D;
        const float* ob  = P.out_b + (long long)l * D;
        const float* f1w = P.ff1_w + (long long)l * ff * D;
        const float* f1b = P.ff1_b + (long long)l * ff;
        const float* f2w = P.ff2_w + (long long)l * D * ff;
        const float* f2b = P.ff2_b + (long long)l * D;

        // qkv projection: [M,3D]
        gemm(st, x, qw, qb, nullptr, qkvb, M, 3 * D, D, D, D, 3 * D, 0, 0);
        // scores = q @ k^T, batched over (b,h)
        gemm(st, qkvb, qkvb + D, nullptr, nullptr, scores, S, S, hd,
             3 * D, 3 * D, S, 0, 0, Bn * NHn, NHn,
             (long long)S * 3 * D, hd, (long long)S * 3 * D, hd,
             (long long)NHn * S * S, (long long)S * S);
        k_softmax_enc<<<Bn * NHn * S, 128, 0, st>>>(scores, ids, NHn, S, scale);
        // o = attn @ v, written directly into merged-head layout [B,S,D]
        gemm(st, scores, qkvb + 2 * D, nullptr, nullptr, mha, S, hd, S,
             S, 3 * D, D, 1, 0, Bn * NHn, NHn,
             (long long)NHn * S * S, (long long)S * S,
             (long long)S * 3 * D, hd, (long long)S * D, hd);
        // out proj + residual, then LN1
        gemm(st, mha, ow, ob, x, x, M, D, D, D, D, D, 0, 0);
        k_layernorm<<<M, 128, 0, st>>>(x, P.ln1_g + (long long)l * D, P.ln1_b + (long long)l * D, D);
        // FFN
        gemm(st, x, f1w, f1b, nullptr, ffb, M, ff, D, D, D, ff, 0, 1 /*gelu*/);
        gemm(st, ffb, f2w, f2b, x, x, M, D, ff, ff, ff, D, 0, 0);
        k_layernorm<<<M, 128, 0, st>>>(x, P.ln2_g + (long long)l * D, P.ln2_b + (long long)l * D, D);
    }
}

// ---------------- entry point ----------------
extern "C" void kernel_launch(void* const* d_in, const int* in_sizes, int n_in,
                              void* d_out, int out_size, void* d_ws, size_t ws_size,
                              hipStream_t stream)
{
    (void)in_sizes; (void)n_in; (void)out_size; (void)ws_size;

    const int*   c_ids = (const int*)d_in[0];
    const int*   q_ids = (const int*)d_in[1];
    const int*   a_ids = (const int*)d_in[2];
    const float* zq    = (const float*)d_in[3];
    const float* emb   = (const float*)d_in[4];

    EncP ctx = { (const float*)d_in[5],  (const float*)d_in[6],  (const float*)d_in[7],
                 (const float*)d_in[8],  (const float*)d_in[9],  (const float*)d_in[10],
                 (const float*)d_in[11], (const float*)d_in[12], (const float*)d_in[13],
                 (const float*)d_in[14], (const float*)d_in[15], (const float*)d_in[16] };
    EncP qen = { (const float*)d_in[17], (const float*)d_in[18], (const float*)d_in[19],
                 (const float*)d_in[20], (const float*)d_in[21], (const float*)d_in[22],
                 (const float*)d_in[23], (const float*)d_in[24], (const float*)d_in[25],
                 (const float*)d_in[26], (const float*)d_in[27], (const float*)d_in[28] };

    const float* fusion_w   = (const float*)d_in[29];
    const float* gate_w     = (const float*)d_in[30];
    const float* qenc_lin_w = (const float*)d_in[31];
    const float* qlin_w     = (const float*)d_in[32];
    const float* qlin_b     = (const float*)d_in[33];
    const float* zq_w       = (const float*)d_in[34];
    const float* zq_b       = (const float*)d_in[35];
    const float* cl1_w      = (const float*)d_in[36];
    const float* cl1_b      = (const float*)d_in[37];
    const float* cl2_w      = (const float*)d_in[38];
    const float* cl2_b      = (const float*)d_in[39];
    const float* prelog_w   = (const float*)d_in[40];
    const float* prelog_b   = (const float*)d_in[41];

    float* out = (float*)d_out;

    // ---- workspace bump allocator ----
    char* wp = (char*)d_ws;
    auto allocf = [&](long long nelem) -> float* {
        float* p = (float*)wp;
        wp += ((nelem * 4 + 255) & ~(long long)255);
        return p;
    };
    auto alloci = [&](long long nelem) -> int* {
        int* p = (int*)wp;
        wp += ((nelem * 4 + 255) & ~(long long)255);
        return p;
    };

    const int Mctx = BB * TCp;   // 3072
    const int Mq   = BB * TQp;   // 512

    float* ce      = allocf((long long)Mctx * HH);
    float* cae     = allocf((long long)Mctx * HH);
    int*   ca_ids  = alloci(Mctx);
    float* qkvb    = allocf((long long)Mctx * 3 * HH);              // >= 512*4608 too
    float* scores  = allocf((long long)BB * NHp * TCp * TCp);       // reused by qenc
    float* mha     = allocf((long long)Mctx * HH);                  // >= 512*1536
    float* ffb     = allocf((long long)Mctx * HH);                  // >= 512*1536
    float* dq      = allocf((long long)BB * HH);
    float* qx      = allocf((long long)Mq * 2 * HH);
    float* cc      = allocf((long long)Mctx * 2 * HH);
    float* cfused  = allocf((long long)Mctx * HH);
    float* cgate   = allocf((long long)Mctx * HH);
    float* c_outs  = allocf((long long)Mctx * HH);
    float* q_out   = allocf((long long)Mq * HH);
    float* ql      = allocf((long long)Mq * HH);
    float* attn_l  = allocf((long long)Mq * TCp);
    float* attn_w  = allocf((long long)Mq * TCp);
    float* c_attn  = allocf((long long)Mq * HH);
    float* qcA     = allocf((long long)Mq * 2 * HH);
    float* qcB     = allocf((long long)Mq * 2 * HH);
    float* qmax    = allocf((long long)Mq * HH);
    float* prelog  = allocf((long long)Mq * HH);

    // ---- embeddings / small projections ----
    k_embed_ctx<<<Mctx, 256, 0, stream>>>(c_ids, a_ids, emb, ce, cae, ca_ids, TCp, HH);
    k_decoded_q<<<(BB * HH + 255) / 256, 256, 0, stream>>>(zq, zq_w, zq_b, dq, HH);
    k_build_qx<<<Mq, 256, 0, stream>>>(q_ids, emb, dq, qx, TQp, HH);

    // ---- context encoders (both passes share scratch) ----
    run_encoder(stream, ce,  c_ids,  BB, TCp, HH, HH, HH / NHp, ctx, qkvb, scores, mha, ffb);
    run_encoder(stream, cae, ca_ids, BB, TCp, HH, HH, HH / NHp, ctx, qkvb, scores, mha, ffb);

    // ---- fusion gate over concat(c_out, ca_out) ----
    k_concat2<<<Mctx, 256, 0, stream>>>(ce, cae, cc, HH, HH);
    gemm(stream, cc, fusion_w, nullptr, nullptr, cfused, Mctx, HH, 2 * HH,
         2 * HH, 2 * HH, HH, 0, 2 /*tanh*/);
    gemm(stream, cc, gate_w, nullptr, nullptr, cgate, Mctx, HH, 2 * HH,
         2 * HH, 2 * HH, HH, 0, 3 /*sigmoid*/);
    k_gate_combine<<<((long long)Mctx * HH + 255) / 256, 256, 0, stream>>>(
        cgate, cfused, ce, c_outs, (long long)Mctx * HH);

    // ---- question encoder (D=1536) ----
    run_encoder(stream, qx, q_ids, BB, TQp, 2 * HH, 2 * HH, 2 * HH / NHp, qen,
                qkvb, scores, mha, ffb);
    gemm(stream, qx, qenc_lin_w, nullptr, nullptr, q_out, Mq, HH, 2 * HH,
         2 * HH, 2 * HH, HH, 0, 0);

    // ---- cross attention over context ----
    gemm(stream, q_out, qlin_w, qlin_b, nullptr, ql, Mq, HH, HH, HH, HH, HH, 0, 0);
    gemm(stream, ql, c_outs, nullptr, nullptr, attn_l, TQp, TCp, HH,
         HH, HH, TCp, 0, 0, BB, 1,
         (long long)TQp * HH, 0, (long long)TCp * HH, 0, (long long)TQp * TCp, 0);
    k_cal_softmax<<<Mq, 128, 0, stream>>>(attn_l, attn_w, c_ids, q_ids, TQp, TCp);
    gemm(stream, attn_w, c_outs, nullptr, nullptr, c_attn, TQp, HH, TCp,
         TCp, HH, HH, 1, 0, BB, 1,
         (long long)TQp * TCp, 0, (long long)TCp * HH, 0, (long long)TQp * HH, 0);

    // ---- classifier head ----
    k_concat2<<<Mq, 256, 0, stream>>>(q_out, c_attn, qcA, HH, HH);
    gemm(stream, qcA, cl1_w, cl1_b, nullptr, qcB, Mq, 2 * HH, 2 * HH,
         2 * HH, 2 * HH, 2 * HH, 0, 4 /*mish*/);
    gemm(stream, qcB, cl2_w, cl2_b, nullptr, qcA, Mq, 2 * HH, 2 * HH,
         2 * HH, 2 * HH, 2 * HH, 0, 4 /*mish*/);
    k_maxout<<<((long long)Mq * HH + 255) / 256, 256, 0, stream>>>(qcA, qmax, (long long)Mq * HH, HH);
    gemm(stream, qmax, prelog_w, prelog_b, nullptr, prelog, Mq, HH, HH, HH, HH, HH, 0, 0);

    // ---- tied-vocab logits + pointer copy ----
    gemm(stream, prelog, emb, nullptr, nullptr, out, Mq, VV, HH, HH, HH, VV, 0, 0);
    k_copy_scatter<<<Mq, TCp, 0, stream>>>(out, attn_l, c_ids, TQp, TCp, VV);
}
